// ODE_func_mix_autoencoder_46926812677055
// MI455X (gfx1250) — matrix operand order repaired and checked by reading, not compiled, not run on hardware
//
#include <hip/hip_runtime.h>
#include <stdint.h>

// Problem constants (from the reference)
#define NTOT   40000          // N*V nodes
#define NEDGE  640000
#define CDIM   64
#define KTOT   27
#define KA     (KTOT*CDIM)    // 1728  (scatter K-dim)
#define KBTOT  (KA + CDIM)    // 1792  (+root-weight K-dim)
#define NKB    (KBTOT/32)     // 56 K-blocks of 32 for WMMA bf16
#define NKB_A  (KA/32)        // 54 blocks belonging to the scatter part
#define NMT    (NTOT/16)      // 2500 M tiles

typedef __bf16        bf16x16 __attribute__((ext_vector_type(16)));
typedef float         f32x8   __attribute__((ext_vector_type(8)));
typedef unsigned int  u32x4   __attribute__((ext_vector_type(4)));

union ABCast { u32x4 q[2]; bf16x16 v; };

__device__ __forceinline__ float eluf(float v) {
    return v > 0.f ? v : (__expf(v) - 1.f);
}
// round-to-nearest-even pack of two f32 into a dword of two bf16
__device__ __forceinline__ unsigned int bf16pk(float a, float b) {
    unsigned int ua = __float_as_uint(a), ub = __float_as_uint(b);
    ua += 0x7FFFu + ((ua >> 16) & 1u);
    ub += 0x7FFFu + ((ub >> 16) & 1u);
    return (ua >> 16) | (ub & 0xFFFF0000u);
}

// ---------------------------------------------------------------- zero fill
__global__ void k_fill_zero(u32x4* __restrict__ p, long long n16) {
    long long i = (long long)blockIdx.x * blockDim.x + threadIdx.x;
    long long stride = (long long)gridDim.x * blockDim.x;
    u32x4 z = {0u, 0u, 0u, 0u};
    for (; i < n16; i += stride) p[i] = z;
}

// ---------------------------------------------------------------- encoder
__global__ void __launch_bounds__(256) k_encoder(
    const float* __restrict__ x,
    const float* __restrict__ w0, const float* __restrict__ b0,
    const float* __restrict__ w1, const float* __restrict__ b1,
    const float* __restrict__ w2, const float* __restrict__ b2,
    float* __restrict__ hf, unsigned int* __restrict__ hb)
{
    __shared__ float sw0[32], sb0[16], sw1[512], sb1[32], sw2[2048], sb2[64];
    for (int i = threadIdx.x; i < 32;   i += 256) sw0[i] = w0[i];
    for (int i = threadIdx.x; i < 16;   i += 256) sb0[i] = b0[i];
    for (int i = threadIdx.x; i < 512;  i += 256) sw1[i] = w1[i];
    for (int i = threadIdx.x; i < 32;   i += 256) sb1[i] = b1[i];
    for (int i = threadIdx.x; i < 2048; i += 256) sw2[i] = w2[i];
    for (int i = threadIdx.x; i < 64;   i += 256) sb2[i] = b2[i];
    __syncthreads();
    int node = blockIdx.x * 256 + threadIdx.x;
    if (node >= NTOT) return;
    float x0 = x[node*2+0], x1 = x[node*2+1];
    float h0[16];
    #pragma unroll
    for (int o = 0; o < 16; ++o)
        h0[o] = eluf(sw0[o*2+0]*x0 + sw0[o*2+1]*x1 + sb0[o]);
    float h1[32];
    #pragma unroll
    for (int o = 0; o < 32; ++o) {
        float s = sb1[o];
        #pragma unroll
        for (int i = 0; i < 16; ++i) s += sw1[o*16+i] * h0[i];
        h1[o] = eluf(s);
    }
    for (int o = 0; o < 64; o += 2) {
        float s0 = sb2[o], s1 = sb2[o+1];
        #pragma unroll
        for (int i = 0; i < 32; ++i) {
            s0 += sw2[o*32+i]     * h1[i];
            s1 += sw2[(o+1)*32+i] * h1[i];
        }
        s0 = eluf(s0); s1 = eluf(s1);
        hf[node*64 + o]     = s0;
        hf[node*64 + o + 1] = s1;
        hb[node*32 + (o>>1)] = bf16pk(s0, s1);   // bf16 copy for WMMA root part
    }
}

// ------------------------------------------------- pack B into WMMA layout
// B[K][n] rows: K<1728 -> sp_w (already [K,64]); K>=1728 -> sp_root[K-1728][64].
// Packed per (ntile, kblock): lane -> n = nt*16+lane%16, khalf = lane/16;
// vgpr i holds K = kb*32 + khalf*8 + (i/4)*16 + 2*(i%4) (+1) packed in one dword.
__global__ void k_packB(const float* __restrict__ spw,
                        const float* __restrict__ sproot,
                        unsigned int* __restrict__ bp)
{
    int idx = blockIdx.x * 256 + threadIdx.x;
    if (idx >= 4*NKB*32*8) return;
    int i    = idx & 7;
    int lane = (idx >> 3) & 31;
    int blk  = idx >> 8;           // nt*NKB + kb
    int kb   = blk % NKB;
    int nt   = blk / NKB;
    int n  = nt*16 + (lane & 15);
    int kq = kb*32 + (lane >> 4)*8 + (i >> 2)*16 + 2*(i & 3);   // always even
    float f0, f1;
    if (kq < KA) { f0 = spw[kq*64 + n];       f1 = spw[(kq+1)*64 + n]; }
    else { int r = kq - KA; f0 = sproot[r*64 + n]; f1 = sproot[(r+1)*64 + n]; }
    bp[idx] = bf16pk(f0, f1);
}

// -------------------------------------------- edge scatter (one wave/edge)
// acc kept in bf16 (138 MB -> L2-resident) via global_atomic_pk_add_bf16.
__global__ void __launch_bounds__(256) k_scatter(
    const int* __restrict__ ei, const float* __restrict__ ea,
    const float* __restrict__ hf,
    unsigned long long accBase, unsigned long long degBase)
{
    int lane = threadIdx.x & 31;
    int e = blockIdx.x * 8 + (threadIdx.x >> 5);
    if (e >= NEDGE) return;
    int src = ei[e];
    int dst = ei[NEDGE + e];
    float v0 = ea[e*3+0]*2.f, v1 = ea[e*3+1]*2.f, v2 = ea[e*3+2]*2.f;
    float k0 = fminf(fmaxf(floorf(v0), 0.f), 1.f);
    float k1 = fminf(fmaxf(floorf(v1), 0.f), 1.f);
    float k2 = fminf(fmaxf(floorf(v2), 0.f), 1.f);
    float f0 = v0 - k0, f1 = v1 - k1, f2 = v2 - k2;
    int i0 = (int)k0, i1 = (int)k1, i2 = (int)k2;
    const float2* hf2 = (const float2*)hf;
    float2 cc = hf2[src*32 + lane];            // 2 channels per lane
    if (lane == 0) {
        unsigned int doff = (unsigned int)dst * 4u;
        float one = 1.f;
        asm volatile("global_atomic_add_f32 %0, %1, %2"
                     :: "v"(doff), "v"(one), "s"(degBase) : "memory");
    }
    unsigned int chan = (unsigned int)lane * 2u;
    #pragma unroll
    for (int s = 0; s < 8; ++s) {
        int b0 = s & 1, b1 = (s >> 1) & 1, b2 = (s >> 2) & 1;
        float w = (b0 ? f0 : 1.f - f0) * (b1 ? f1 : 1.f - f1) * (b2 ? f2 : 1.f - f2);
        int flat = (i0+b0)*9 + (i1+b1)*3 + (i2+b2);
        unsigned int off = ((unsigned int)dst * (unsigned)KA
                            + (unsigned)flat * 64u + chan) * 2u;  // bytes (bf16)
        unsigned int data = bf16pk(w * cc.x, w * cc.y);
        asm volatile("global_atomic_pk_add_bf16 %0, %1, %2"
                     :: "v"(off), "v"(data), "s"(accBase) : "memory");
    }
}

// -------------------------------- spline GEMM: one wave -> 16 rows x 64 cols
// Double-buffered: A/B fragments for block kb+1 are prefetched before the
// four WMMAs of block kb so loads overlap matrix ops (partial loadcnt waits).
__global__ void __launch_bounds__(256) k_spline_gemm(
    const u32x4* __restrict__ accq, const u32x4* __restrict__ hbq,
    const u32x4* __restrict__ bpq,  const float* __restrict__ deg,
    const float* __restrict__ bias, float* __restrict__ h2)
{
    int lane = threadIdx.x & 31;
    int wt = blockIdx.x * 8 + (threadIdx.x >> 5);
    if (wt >= NMT) return;                       // uniform per wave
    int m0   = wt * 16;
    int half = lane >> 4;
    int rowA = m0 + (lane & 15);
    f32x8 acc[4] = {};
    // A row: 1728 bf16 = 216 uint4; per kblock two b128 at K-offsets half*8, 16+half*8
    const u32x4* aline = accq + (size_t)rowA * 216 + half;
    const u32x4* bbase = bpq + (size_t)lane * 2;

    ABCast a; u32x4 b[4][2];
    a.q[0] = __builtin_nontemporal_load(&aline[0]);
    a.q[1] = __builtin_nontemporal_load(&aline[2]);
    #pragma unroll
    for (int nt = 0; nt < 4; ++nt) {
        const u32x4* bb = bbase + (size_t)(nt*NKB) * 64;
        b[nt][0] = bb[0]; b[nt][1] = bb[1];
    }
    #pragma unroll 2
    for (int kb = 0; kb < NKB_A; ++kb) {
        // prefetch block kb+1 (dummy re-read of block 0 on the last iteration)
        int kn = (kb + 1 < NKB_A) ? kb + 1 : 0;
        ABCast an; u32x4 bn[4][2];
        an.q[0] = __builtin_nontemporal_load(&aline[kn*4]);
        an.q[1] = __builtin_nontemporal_load(&aline[kn*4 + 2]);
        #pragma unroll
        for (int nt = 0; nt < 4; ++nt) {
            const u32x4* bb = bbase + (size_t)(nt*NKB + kn) * 64;
            bn[nt][0] = bb[0]; bn[nt][1] = bb[1];
        }
        // compute with current buffers
        #pragma unroll
        for (int nt = 0; nt < 4; ++nt) {
            ABCast bc; bc.q[0] = b[nt][0]; bc.q[1] = b[nt][1];
            acc[nt] = __builtin_amdgcn_wmma_f32_16x16x32_bf16(
                false, a.v, false, bc.v, (short)0, acc[nt], false, false);
        }
        a = an;
        #pragma unroll
        for (int nt = 0; nt < 4; ++nt) { b[nt][0] = bn[nt][0]; b[nt][1] = bn[nt][1]; }
    }
    // degree normalization applies ONLY to the scatter contraction
    float ds[8];
    #pragma unroll
    for (int j = 0; j < 8; ++j)
        ds[j] = 1.f / fmaxf(deg[m0 + j + 8*half], 1.f);
    #pragma unroll
    for (int nt = 0; nt < 4; ++nt)
        #pragma unroll
        for (int j = 0; j < 8; ++j)
            acc[nt][j] *= ds[j];
    // root-weight part: A = encoder features (bf16), K blocks 54..55
    const u32x4* hline = hbq + (size_t)rowA * 8 + half;
    #pragma unroll
    for (int kk = 0; kk < 2; ++kk) {
        ABCast ar; ar.q[0] = hline[kk*4]; ar.q[1] = hline[kk*4 + 2];
        int kb = NKB_A + kk;
        #pragma unroll
        for (int nt = 0; nt < 4; ++nt) {
            const u32x4* bb = bbase + (size_t)(nt*NKB + kb) * 64;
            ABCast bc; bc.q[0] = bb[0]; bc.q[1] = bb[1];
            acc[nt] = __builtin_amdgcn_wmma_f32_16x16x32_bf16(
                false, ar.v, false, bc.v, (short)0, acc[nt], false, false);
        }
    }
    // epilogue: + bias, ELU, store (C/D layout: vgpr j -> row j+8*half, lane -> col)
    #pragma unroll
    for (int nt = 0; nt < 4; ++nt) {
        int n = nt*16 + (lane & 15);
        float bn = bias[n];
        #pragma unroll
        for (int j = 0; j < 8; ++j) {
            int m = m0 + j + 8*half;
            h2[(size_t)m*64 + n] = eluf(acc[nt][j] + bn);
        }
    }
}

// ---------------------------------------------------------------- decoder
__global__ void __launch_bounds__(256) k_decoder(
    const float* __restrict__ h2,
    const float* __restrict__ w0, const float* __restrict__ b0,
    const float* __restrict__ w1, const float* __restrict__ b1,
    const float* __restrict__ w2, const float* __restrict__ b2,
    float* __restrict__ out)
{
    __shared__ float sw0[2048], sb0[32], sw1[512], sb1[16], sw2[32], sb2[2];
    for (int i = threadIdx.x; i < 2048; i += 256) sw0[i] = w0[i];
    for (int i = threadIdx.x; i < 32;   i += 256) sb0[i] = b0[i];
    for (int i = threadIdx.x; i < 512;  i += 256) sw1[i] = w1[i];
    for (int i = threadIdx.x; i < 16;   i += 256) sb1[i] = b1[i];
    for (int i = threadIdx.x; i < 32;   i += 256) sw2[i] = w2[i];
    for (int i = threadIdx.x; i < 2;    i += 256) sb2[i] = b2[i];
    __syncthreads();
    int node = blockIdx.x * 256 + threadIdx.x;
    if (node >= NTOT) return;
    float hin[64];
    #pragma unroll
    for (int i = 0; i < 64; ++i) hin[i] = h2[node*64 + i];
    float a0[32];
    #pragma unroll
    for (int o = 0; o < 32; ++o) {
        float s = sb0[o];
        #pragma unroll
        for (int i = 0; i < 64; ++i) s += sw0[o*64+i] * hin[i];
        a0[o] = eluf(s);
    }
    float a1[16];
    #pragma unroll
    for (int o = 0; o < 16; ++o) {
        float s = sb1[o];
        #pragma unroll
        for (int i = 0; i < 32; ++i) s += sw1[o*32+i] * a0[i];
        a1[o] = eluf(s);
    }
    float o0 = sb2[0], o1 = sb2[1];
    #pragma unroll
    for (int i = 0; i < 16; ++i) { o0 += sw2[i] * a1[i]; o1 += sw2[16+i] * a1[i]; }
    out[node*2+0] = tanhf(o0);
    out[node*2+1] = tanhf(o1);
}

// ---------------------------------------------------------------- launcher
extern "C" void kernel_launch(void* const* d_in, const int* in_sizes, int n_in,
                              void* d_out, int out_size, void* d_ws, size_t ws_size,
                              hipStream_t stream)
{
    (void)in_sizes; (void)n_in; (void)out_size; (void)ws_size;
    const float* x    = (const float*)d_in[1];
    const int*   ei   = (const int*)  d_in[2];
    const float* ea   = (const float*)d_in[3];
    const float* ew0  = (const float*)d_in[4];
    const float* eb0  = (const float*)d_in[5];
    const float* ew1  = (const float*)d_in[6];
    const float* eb1  = (const float*)d_in[7];
    const float* ew2  = (const float*)d_in[8];
    const float* eb2  = (const float*)d_in[9];
    const float* spw  = (const float*)d_in[10];
    const float* sprt = (const float*)d_in[11];
    const float* spb  = (const float*)d_in[12];
    const float* dw0  = (const float*)d_in[13];
    const float* db0  = (const float*)d_in[14];
    const float* dw1  = (const float*)d_in[15];
    const float* db1  = (const float*)d_in[16];
    const float* dw2  = (const float*)d_in[17];
    const float* db2  = (const float*)d_in[18];

    char* ws = (char*)d_ws;
    const size_t ACC_B = (size_t)NTOT * KA * 2;      // 138,240,000  (bf16 buckets)
    const size_t DEG_B = (size_t)NTOT * 4;           //     160,000
    const size_t HF_B  = (size_t)NTOT * CDIM * 4;    //  10,240,000
    const size_t HB_B  = (size_t)NTOT * CDIM * 2;    //   5,120,000
    const size_t BP_B  = (size_t)KBTOT * CDIM * 2;   //     229,376
    char* p_acc = ws;
    char* p_deg = p_acc + ACC_B;
    char* p_hf  = p_deg + DEG_B;
    char* p_hb  = p_hf  + HF_B;
    char* p_bp  = p_hb  + HB_B;
    char* p_h2  = p_bp  + BP_B;

    long long n16 = (long long)((ACC_B + DEG_B) / 16);
    k_fill_zero<<<2048, 256, 0, stream>>>((u32x4*)p_acc, n16);
    k_encoder<<<(NTOT + 255)/256, 256, 0, stream>>>(
        x, ew0, eb0, ew1, eb1, ew2, eb2, (float*)p_hf, (unsigned int*)p_hb);
    k_packB<<<(4*NKB*32*8 + 255)/256, 256, 0, stream>>>(
        spw, sprt, (unsigned int*)p_bp);
    k_scatter<<<NEDGE/8, 256, 0, stream>>>(
        ei, ea, (const float*)p_hf,
        (unsigned long long)(uintptr_t)p_acc,
        (unsigned long long)(uintptr_t)p_deg);
    k_spline_gemm<<<(NMT + 7)/8, 256, 0, stream>>>(
        (const u32x4*)p_acc, (const u32x4*)p_hb, (const u32x4*)p_bp,
        (const float*)p_deg, spb, (float*)p_h2);
    k_decoder<<<(NTOT + 255)/256, 256, 0, stream>>>(
        (const float*)p_h2, dw0, db0, dw1, db1, dw2, db2, (float*)d_out);
}
